// GraphConvolutionBlock_90744069030060
// MI455X (gfx1250) — matrix operand order repaired and checked by reading, not compiled, not run on hardware
//
#include <hip/hip_runtime.h>

// ---------------------------------------------------------------------------
// MI455X (gfx1250) relational GCN block.
//   out[b,i,o] = cb + sum_r cw[r] * sum_j adjn[r,i,j] *
//                relu( sum_d draft[b,j,d]*W[r,o,d] + b[r,o] )
// Strategy: f16 WMMA (fp32 accum) for the 86-GFLOP GEMM; per-WG tile of 8
// batches (80 rows); A staged in LDS as f16; W pre-converted to f16 and
// streamed from L2; tiny 10x10 adj mixing done with VALU FMAs.
// ---------------------------------------------------------------------------

typedef __attribute__((ext_vector_type(16))) _Float16 v16h;
typedef __attribute__((ext_vector_type(8)))  _Float16 v8h;
typedef __attribute__((ext_vector_type(4)))  _Float16 v4h;
typedef __attribute__((ext_vector_type(8)))  float    v8f;
typedef __attribute__((ext_vector_type(4)))  float    v4f;

#define LDA 264   // f16 stride for staged draft tile (256 + 8 pad)
#define LDH 264   // f32 stride for H tile (256 + 8 pad)
#define SMEM_A_BYTES (80 * LDA * 2)            // 42240
#define SMEM_H_BYTES (80 * LDH * 4)            // 84480
#define SMEM_C_BYTES (400 * 4)                 // 1600
#define SMEM_BYTES   (SMEM_A_BYTES + SMEM_H_BYTES + SMEM_C_BYTES)

// --- setup: W fp32 -> f16 (row-major (r,o,d), same layout) -----------------
__global__ __launch_bounds__(256) void wcvt_kernel(const float* __restrict__ W,
                                                   _Float16* __restrict__ W16,
                                                   int n4) {
  int i = blockIdx.x * blockDim.x + threadIdx.x;
  if (i < n4) {
    v4f w = ((const v4f*)W)[i];
    v4h h;
#pragma unroll
    for (int e = 0; e < 4; ++e) h[e] = (_Float16)w[e];
    ((v4h*)W16)[i] = h;
  }
}

// --- setup: coef[r][i][j] = cw[r] * adj[r,i,j] / max(sum_j |adj[r,i,j]|, eps)
__global__ __launch_bounds__(64) void coef_kernel(const float* __restrict__ adj,
                                                  const float* __restrict__ cw,
                                                  float* __restrict__ coef) {
  int t = threadIdx.x;
  if (t < 40) {
    int r = t / 10;
    const float* row = adj + t * 10;
    float s = 0.f;
#pragma unroll
    for (int j = 0; j < 10; ++j) s += fabsf(row[j]);
    float scale = cw[r] / fmaxf(s, 1e-12f);
#pragma unroll
    for (int j = 0; j < 10; ++j) coef[t * 10 + j] = row[j] * scale;
  }
}

// --- main kernel: one WG handles 8 batches (80 rows x 256) -----------------
__global__ __launch_bounds__(256) void rgcn_kernel(
    const float* __restrict__ draft, const _Float16* __restrict__ W16,
    const float* __restrict__ bias, const float* __restrict__ coef,
    const float* __restrict__ cbp, float* __restrict__ out) {
  extern __shared__ __align__(16) unsigned char smem[];
  _Float16* sA    = (_Float16*)smem;                               // 80 x LDA f16
  float*    sH    = (float*)(smem + SMEM_A_BYTES);                 // 80 x LDH f32
  float*    sCoef = (float*)(smem + SMEM_A_BYTES + SMEM_H_BYTES);  // 4x10x10

  const int tid   = threadIdx.x;
  const int lane  = tid & 31;
  const int wave  = tid >> 5;    // 8 waves; wave w owns batch w for mixing
  const int lhalf = lane >> 4;   // lane-half select (ISA 16-bit A/B layouts)
  const int l16   = lane & 15;
  const int wg    = blockIdx.x;

  // stage draft tile -> LDS f16 (coalesced 16B global loads, 8B LDS stores)
  {
    const v4f* src = (const v4f*)(draft + (size_t)wg * 80 * 256);
    for (int idx = tid; idx < 80 * 64; idx += 256) {
      int row = idx >> 6;
      int c4  = idx & 63;
      v4f v = src[row * 64 + c4];
      v4h h;
#pragma unroll
      for (int e = 0; e < 4; ++e) h[e] = (_Float16)v[e];
      *(v4h*)(sA + row * LDA + c4 * 4) = h;
    }
  }
  if (tid < 400) sCoef[tid] = coef[tid];
  __syncthreads();

  float acc[10][8];
#pragma unroll
  for (int i = 0; i < 10; ++i)
#pragma unroll
    for (int c = 0; c < 8; ++c) acc[i][c] = 0.f;

  const int c0 = lane * 8;  // mixing-phase column base (lane owns 8 cols)

  for (int r = 0; r < 4; ++r) {
    // ===== GEMM: H = relu(draft_tile @ W_r^T + b_r), H is 80x256 ==========
#pragma unroll 1
    for (int nt2 = 0; nt2 < 2; ++nt2) {
      const int nt = wave * 2 + nt2;       // this wave's N-tile
      const int o  = nt * 16 + l16;        // output column this lane holds
      const v8f zero = {};
      v8f c[5];
#pragma unroll
      for (int mt = 0; mt < 5; ++mt) c[mt] = zero;

      // B-operand: lane n holds W row o, 16 contiguous d at k0 + 16*lhalf
      const _Float16* wrow = W16 + ((size_t)(r * 256 + o)) * 256 + lhalf * 16;
#pragma unroll
      for (int k = 0; k < 8; ++k) {
        v8h blo = *(const v8h*)(wrow + k * 32);
        v8h bhi = *(const v8h*)(wrow + k * 32 + 8);
        v16h bm = __builtin_shufflevector(blo, bhi, 0, 1, 2, 3, 4, 5, 6, 7, 8,
                                          9, 10, 11, 12, 13, 14, 15);
#pragma unroll
        for (int mt = 0; mt < 5; ++mt) {
          // A-operand: lane m=l16 row, 16 contiguous K at k0 + 16*lhalf
          const _Float16* arow =
              sA + (mt * 16 + l16) * LDA + k * 32 + lhalf * 16;
          v8h alo = *(const v8h*)(arow);
          v8h ahi = *(const v8h*)(arow + 8);
          v16h am = __builtin_shufflevector(alo, ahi, 0, 1, 2, 3, 4, 5, 6, 7,
                                            8, 9, 10, 11, 12, 13, 14, 15);
          c[mt] = __builtin_amdgcn_wmma_f32_16x16x32_f16(
              false, am, false, bm, (short)0, c[mt], false, false);
        }
      }
      // bias + ReLU, spill C tiles to LDS (C elem e -> row mt*16 + 8*lhalf + e)
      const float bv = bias[r * 256 + o];
#pragma unroll
      for (int mt = 0; mt < 5; ++mt)
#pragma unroll
        for (int e = 0; e < 8; ++e) {
          int row = mt * 16 + lhalf * 8 + e;
          sH[row * LDH + o] = fmaxf(c[mt][e] + bv, 0.f);
        }
    }
    __syncthreads();

    // ===== adj mixing: acc[i][*] += coef[r,i,j] * H[wave*10+j][*] =========
    const float* crow = sCoef + r * 100;
#pragma unroll
    for (int j = 0; j < 10; ++j) {
      const float* hp = sH + (wave * 10 + j) * LDH + c0;
      v4f h0 = *(const v4f*)hp;
      v4f h1 = *(const v4f*)(hp + 4);
#pragma unroll
      for (int i = 0; i < 10; ++i) {
        float f = crow[i * 10 + j];
#pragma unroll
        for (int e = 0; e < 4; ++e) {
          acc[i][e]     = fmaf(f, h0[e], acc[i][e]);
          acc[i][e + 4] = fmaf(f, h1[e], acc[i][e + 4]);
        }
      }
    }
    __syncthreads();
  }

  // ===== epilogue: + cb, write (b,i,o) fp32, coalesced 16B stores =========
  const float cbv = cbp[0];
  float* obase = out + (size_t)(wg * 8 + wave) * 10 * 256 + c0;
#pragma unroll
  for (int i = 0; i < 10; ++i) {
    v4f o0, o1;
#pragma unroll
    for (int e = 0; e < 4; ++e) {
      o0[e] = acc[i][e] + cbv;
      o1[e] = acc[i][e + 4] + cbv;
    }
    *(v4f*)(obase + i * 256)     = o0;
    *(v4f*)(obase + i * 256 + 4) = o1;
  }
}

extern "C" void kernel_launch(void* const* d_in, const int* in_sizes, int n_in,
                              void* d_out, int out_size, void* d_ws,
                              size_t ws_size, hipStream_t stream) {
  const float* draft = (const float*)d_in[0];  // (B,10,256)
  const float* adj   = (const float*)d_in[1];  // (4,10,10)
  const float* W     = (const float*)d_in[2];  // (4,256,256)
  const float* b     = (const float*)d_in[3];  // (4,256)
  const float* cw    = (const float*)d_in[4];  // (4,)
  const float* cb    = (const float*)d_in[5];  // (1,)
  float* out = (float*)d_out;                  // (B,10,256)

  const int B = in_sizes[0] / (10 * 256);

  // workspace layout: [ W16: 4*256*256 f16 | coef: 400 f32 ]
  _Float16* W16 = (_Float16*)d_ws;
  float* coef = (float*)((char*)d_ws + (size_t)4 * 256 * 256 * sizeof(_Float16));

  const int n4 = 4 * 256 * 256 / 4;
  wcvt_kernel<<<(n4 + 255) / 256, 256, 0, stream>>>(W, W16, n4);
  coef_kernel<<<1, 64, 0, stream>>>(adj, cw, coef);
  rgcn_kernel<<<B / 8, 256, SMEM_BYTES, stream>>>(draft, W16, b, coef, cb, out);
}
